// VideoFeatureFusion2_69715909149184
// MI455X (gfx1250) — matrix operand order repaired and checked by reading, not compiled, not run on hardware
//
#include <hip/hip_runtime.h>
#include <math.h>

// ---------------------------------------------------------------------------
// Types for CDNA5 WMMA (wave32): v_wmma_f32_16x16x32_bf16
// ---------------------------------------------------------------------------
typedef __attribute__((ext_vector_type(16))) __bf16 v16bf;
typedef __attribute__((ext_vector_type(8)))  float  v8f;

__device__ inline __bf16 f2bf(float f) { return (__bf16)f; }  // native cvt

__device__ inline float sigm(float x) { return 1.0f / (1.0f + __expf(-x)); }
__device__ inline int clampi(int v, int lo, int hi) { return v < lo ? lo : (v > hi ? hi : v); }

// ---------------------------------------------------------------------------
// Unified implicit-GEMM convolution using bf16 WMMA.
//   GEMM: M=Cout, N=B*Ho*Wo, K=Cin*kh*kw.
// Block = 128 threads = 4 waves. Block tile: 64(M) x 32(N). Each wave owns a
// 16(M)x32(N) tile -> one A fragment, two B fragments, two WMMAs per K-step
// of 32. B tile staged in LDS **N-major** (Bs[n][k]) so each lane's fragment
// is one contiguous 32B run -> 2x ds_load_b128 per fragment. Each thread owns
// a fixed N column (loop-invariant (b,ho,wo) decode) and 8 K-slots whose
// (cin, rr) advance incrementally (no div/mod in the K loop); (ky*dil,kx*dil)
// comes from a small LDS table. All loader index math is 32-bit.
// When om != null the loader performs the modulated-deformable bilinear
// sample (y-off ch [0,kk), x-off [kk,2kk), mask logits [2kk,3kk)).
// Epilogue fuses bias, batchnorm, relu, residual add.
// ---------------------------------------------------------------------------
__global__ __launch_bounds__(128)
void conv2d_wmma_kernel(const float* __restrict__ x,
                        const float* __restrict__ om,       // deform offsets/mask or null
                        const float* __restrict__ w,        // [Cout, Cin*kh*kw]
                        const float* __restrict__ bias,     // [Cout] or null
                        const float* __restrict__ bn_g, const float* __restrict__ bn_b,
                        const float* __restrict__ bn_m, const float* __restrict__ bn_v,
                        const float* __restrict__ residual, // same shape as out, or null
                        float* __restrict__ out,
                        int B, int Cin, int H, int W,
                        int Cout, int kh, int kw,
                        int stride, int pad, int dil,
                        int Ho, int Wo, int relu)
{
  __shared__ alignas(32) __bf16 Bs[32][32];   // [N][K] (N-major!)
  __shared__ int kyx[64];                     // rr -> (ky*dil)<<16 | (kx*dil)

  const int tid  = threadIdx.x;
  const int lane = tid & 31;
  const int hi16 = lane >> 4;
  const int l16  = lane & 15;

  const int HoWo = Ho * Wo;
  const int NN   = B * HoWo;
  const int n_base = blockIdx.x * 32;
  const int m_base = (blockIdx.y * 4 + (tid >> 5)) * 16;
  const int khw  = kh * kw;
  const int K    = Cin * khw;
  const int HW   = H * W;

  if (tid < khw) kyx[tid] = (((tid / kw) * dil) << 16) | ((tid % kw) * dil);

  // ---- loop-invariant N-column decode (one per thread)
  const int nn = tid & 31;
  const int kslot0 = tid >> 5;               // 0..3; slot i covers K-row kslot0+4i
  const int n_mine = n_base + nn;
  const bool nvalid = (n_mine < NN);
  int bq = 0, ho = 0, wo = 0;
  if (nvalid) {
    bq = n_mine / HoWo;
    int r = n_mine - bq * HoWo;
    ho = r / Wo;
    wo = r - ho * Wo;
  }
  const int oy = ho * stride - pad;
  const int ox = wo * stride - pad;
  const float* xb_base = x + (size_t)bq * Cin * HW;              // 64-bit once
  const float* omt = om ? om + (size_t)bq * 3 * khw * HoWo + ho * Wo + wo : nullptr;

  // ---- incremental (cin, rr) state for the 8 K-slots
  const int dc = 32 / khw;
  const int dr = 32 - dc * khw;
  int cin_s[8], rr_s[8];
#pragma unroll
  for (int i = 0; i < 8; ++i) {
    int kk = kslot0 + 4 * i;
    cin_s[i] = kk / khw;
    rr_s[i]  = kk - cin_s[i] * khw;
  }

  v8f acc0, acc1;
#pragma unroll
  for (int i = 0; i < 8; ++i) { acc0[i] = 0.0f; acc1[i] = 0.0f; }

  for (int k0 = 0; k0 < K; k0 += 32) {
    __syncthreads();
    // ---- stage 32(K) x 32(N) B tile, N-major
#pragma unroll
    for (int i = 0; i < 8; ++i) {
      const int kk = kslot0 + 4 * i;
      float val = 0.0f;
      if (nvalid && (k0 + kk) < K) {
        const int cin = cin_s[i];
        const int rr  = rr_s[i];
        const int pk  = kyx[rr];
        const int kyd = pk >> 16, kxd = pk & 0xffff;
        if (om == nullptr) {
          int iy = oy + kyd;
          int ix = ox + kxd;
          if ((unsigned)iy < (unsigned)H && (unsigned)ix < (unsigned)W)
            val = xb_base[cin * HW + iy * W + ix];               // 32-bit offsets
        } else {
          float py = omt[rr * HoWo]             + (float)(oy + kyd);
          float px = omt[(khw + rr) * HoWo]     + (float)(ox + kxd);
          float ms = sigm(omt[(2 * khw + rr) * HoWo]);
          float y0f = floorf(py), x0f = floorf(px);
          float wy = py - y0f, wx = px - x0f;
          int y0 = (int)y0f, x0 = (int)x0f;
          const float* xb = xb_base + cin * HW;
          int i00 = y0 * W + x0;
          float v00 = ((unsigned)y0     < (unsigned)H && (unsigned)x0     < (unsigned)W) ? xb[i00]         : 0.0f;
          float v01 = ((unsigned)y0     < (unsigned)H && (unsigned)(x0+1) < (unsigned)W) ? xb[i00 + 1]     : 0.0f;
          float v10 = ((unsigned)(y0+1) < (unsigned)H && (unsigned)x0     < (unsigned)W) ? xb[i00 + W]     : 0.0f;
          float v11 = ((unsigned)(y0+1) < (unsigned)H && (unsigned)(x0+1) < (unsigned)W) ? xb[i00 + W + 1] : 0.0f;
          val = (v00 * (1.f - wy) * (1.f - wx) + v01 * (1.f - wy) * wx +
                 v10 * wy * (1.f - wx)         + v11 * wy * wx) * ms;
        }
      }
      Bs[nn][kk] = f2bf(val);
      // advance slot state for next K-step (no div/mod)
      cin_s[i] += dc;
      rr_s[i]  += dr;
      if (rr_s[i] >= khw) { rr_s[i] -= khw; cin_s[i] += 1; }
    }
    __syncthreads();

    // ---- A fragment: lanes 0-15 hold K {0..15},
    //      lanes 16-31 hold {16..31}; one contiguous 16-float run.
    v16bf afrag;
    const int m = m_base + l16;
    if (m < Cout && (k0 + 32) <= K) {
      const float* wp = w + (size_t)m * K + k0 + hi16 * 16;
      const float4 a0 = *(const float4*)(wp);
      const float4 a1 = *(const float4*)(wp + 4);
      const float4 a2 = *(const float4*)(wp + 8);
      const float4 a3 = *(const float4*)(wp + 12);
      afrag[0]  = f2bf(a0.x); afrag[1]  = f2bf(a0.y);
      afrag[2]  = f2bf(a0.z); afrag[3]  = f2bf(a0.w);
      afrag[4]  = f2bf(a1.x); afrag[5]  = f2bf(a1.y);
      afrag[6]  = f2bf(a1.z); afrag[7]  = f2bf(a1.w);
      afrag[8]  = f2bf(a2.x); afrag[9]  = f2bf(a2.y);
      afrag[10] = f2bf(a2.z); afrag[11] = f2bf(a2.w);
      afrag[12] = f2bf(a3.x); afrag[13] = f2bf(a3.y);
      afrag[14] = f2bf(a3.z); afrag[15] = f2bf(a3.w);
      __builtin_prefetch(wp + 32, 0, 1);   // next K-step weights -> global_prefetch
    } else if (m < Cout) {
#pragma unroll
      for (int v = 0; v < 8; ++v) {
        int kb = hi16 * 16 + 2 * v;
        int ka = k0 + kb;
        afrag[2 * v]     = f2bf((ka     < K) ? w[(size_t)m * K + ka]     : 0.0f);
        afrag[2 * v + 1] = f2bf((ka + 1 < K) ? w[(size_t)m * K + ka + 1] : 0.0f);
      }
    } else {
#pragma unroll
      for (int j = 0; j < 16; ++j) afrag[j] = f2bf(0.0f);
    }

    // ---- two B fragments: contiguous 32B runs in N-major LDS tile
    const v16bf bf0 = *(const v16bf*)(&Bs[l16][hi16 * 16]);
    const v16bf bf1 = *(const v16bf*)(&Bs[16 + l16][hi16 * 16]);

    acc0 = __builtin_amdgcn_wmma_f32_16x16x32_bf16(
        false, afrag, false, bf0, (short)0, acc0, false, false);
    acc1 = __builtin_amdgcn_wmma_f32_16x16x32_bf16(
        false, afrag, false, bf1, (short)0, acc1, false, false);
  }

  // ---- epilogue: lane holds N=l16 (tile0) / 16+l16 (tile1); VGPR r -> M=r+8*hi16
#pragma unroll
  for (int tile = 0; tile < 2; ++tile) {
    int n = n_base + 16 * tile + l16;
    if (n >= NN) continue;
    int b  = n / HoWo;
    int r0 = n - b * HoWo;
    int hq = r0 / Wo;
    int wq = r0 - hq * Wo;
#pragma unroll
    for (int r = 0; r < 8; ++r) {
      int m = m_base + r + 8 * hi16;
      if (m < Cout) {
        float v = tile ? acc1[r] : acc0[r];
        if (bias) v += bias[m];
        if (bn_g) {
          float s = bn_g[m] * rsqrtf(bn_v[m] + 1e-5f);
          v = v * s + (bn_b[m] - bn_m[m] * s);
        }
        if (relu) v = fmaxf(v, 0.0f);
        size_t oi = ((size_t)(b * Cout + m) * Ho + hq) * Wo + wq;
        if (residual) v += residual[oi];
        out[oi] = v;
      }
    }
  }
}

// ---------------------------------------------------------------------------
// Elementwise / glue kernels
// ---------------------------------------------------------------------------
__global__ void add_kernel(const float* __restrict__ a, const float* __restrict__ b,
                           float* __restrict__ o, long n, int relu)
{
  long i = (long)blockIdx.x * blockDim.x + threadIdx.x;
  if (i >= n) return;
  float v = a[i] + b[i];
  if (relu) v = fmaxf(v, 0.0f);
  o[i] = v;
}

// jax.image.resize bilinear, scale 2 (half-pixel centers, edge clamp)
__global__ void upsample2x_kernel(const float* __restrict__ x, float* __restrict__ y,
                                  int B, int C, int H, int W)
{
  int H2 = 2 * H, W2 = 2 * W;
  long n = (long)B * C * H2 * W2;
  long i = (long)blockIdx.x * blockDim.x + threadIdx.x;
  if (i >= n) return;
  int ow = i % W2; long t = i / W2;
  int oh = t % H2; t /= H2;
  int c  = t % C;  int b = t / C;
  float sy = (oh + 0.5f) * 0.5f - 0.5f;
  float sx = (ow + 0.5f) * 0.5f - 0.5f;
  int y0 = (int)floorf(sy), x0 = (int)floorf(sx);
  float wy = sy - (float)y0, wx = sx - (float)x0;
  const float* xb = x + (size_t)(b * C + c) * H * W;
  int y0c = clampi(y0, 0, H - 1), y1c = clampi(y0 + 1, 0, H - 1);
  int x0c = clampi(x0, 0, W - 1), x1c = clampi(x0 + 1, 0, W - 1);
  float v = xb[(size_t)y0c * W + x0c] * (1.f - wy) * (1.f - wx)
          + xb[(size_t)y0c * W + x1c] * (1.f - wy) * wx
          + xb[(size_t)y1c * W + x0c] * wy * (1.f - wx)
          + xb[(size_t)y1c * W + x1c] * wy * wx;
  y[i] = v;
}

// feats = f_align*sigmoid(alpha) + f_cur*(1-sigmoid(alpha)); in-place on f_cur
__global__ void blend_kernel(const float* __restrict__ fa, float* __restrict__ fc,
                             const float* __restrict__ alpha, int B, int C, int H, int W)
{
  long n = (long)B * C * H * W;
  long i = (long)blockIdx.x * blockDim.x + threadIdx.x;
  if (i >= n) return;
  int w = i % W; long t = i / W;
  int h = t % H; t /= H;
  t /= C; int b = (int)t;
  float a = sigm(alpha[((size_t)b * H + h) * W + w]);
  fc[i] = fa[i] * a + fc[i] * (1.0f - a);
}

__global__ void bn_kernel(float* __restrict__ x,
                          const float* __restrict__ g, const float* __restrict__ be,
                          const float* __restrict__ mn, const float* __restrict__ vr,
                          int B, int C, int HW, int relu)
{
  long n = (long)B * C * HW;
  long i = (long)blockIdx.x * blockDim.x + threadIdx.x;
  if (i >= n) return;
  int c = (int)((i / HW) % C);
  float s = g[c] * rsqrtf(vr[c] + 1e-5f);
  float v = x[i] * s + (be[c] - mn[c] * s);
  if (relu) v = fmaxf(v, 0.0f);
  x[i] = v;
}

// direct transposed conv; wt layout (Cin, Cout, k, k)
__global__ void conv_transpose_kernel(const float* __restrict__ x, const float* __restrict__ wt,
                                      float* __restrict__ y,
                                      int B, int Cin, int H, int W, int Cout,
                                      int k, int stride, int pad)
{
  int Ho = (H - 1) * stride - 2 * pad + k;
  int Wo = (W - 1) * stride - 2 * pad + k;
  long n = (long)B * Cout * Ho * Wo;
  long i = (long)blockIdx.x * blockDim.x + threadIdx.x;
  if (i >= n) return;
  int ow = i % Wo; long t = i / Wo;
  int oh = t % Ho; t /= Ho;
  int co = t % Cout; int b = (int)(t / Cout);
  float acc = 0.0f;
  for (int ky = 0; ky < k; ++ky) {
    int ty = oh + pad - ky;
    if (ty < 0 || (ty % stride) != 0) continue;
    int ih = ty / stride; if (ih >= H) continue;
    for (int kx = 0; kx < k; ++kx) {
      int tx = ow + pad - kx;
      if (tx < 0 || (tx % stride) != 0) continue;
      int iw = tx / stride; if (iw >= W) continue;
      const float* xp = x + (size_t)b * Cin * H * W + (size_t)ih * W + iw;
      const float* wp = wt + ((size_t)co * k + ky) * k + kx;
      for (int ci = 0; ci < Cin; ++ci)
        acc += xp[(size_t)ci * H * W] * wp[(size_t)ci * Cout * k * k];
    }
  }
  y[i] = acc;
}

// ret_off head: tan(sigmoid(mean over kk groups) * 3.14 - 1.57)
__global__ void offset_ret_kernel(const float* __restrict__ om, float* __restrict__ out,
                                  int B, int kkn, int H, int W)
{
  long n = (long)B * 2 * H * W;
  long i = (long)blockIdx.x * blockDim.x + threadIdx.x;
  if (i >= n) return;
  int w = i % W; long t = i / W;
  int h = t % H; t /= H;
  int g = t % 2; int b = (int)(t / 2);
  int HW = H * W;
  const float* base = om + ((size_t)b * 3 * kkn + (size_t)g * kkn) * HW + (size_t)h * W + w;
  float m = 0.0f;
  for (int j = 0; j < kkn; ++j) m += base[(size_t)j * HW];
  m /= (float)kkn;
  out[i] = tanf(sigm(m) * 3.14f - 1.57f);
}

// ---------------------------------------------------------------------------
// Host-side orchestration
// ---------------------------------------------------------------------------
namespace {

struct Bump {
  char* base; size_t off; size_t cap;
  float* alloc(size_t nfloat) {
    size_t bytes = (nfloat * sizeof(float) + 255) & ~(size_t)255;
    float* p = (float*)(base + off);
    off += bytes;
    return p;
  }
};

inline long cdivl(long a, long b) { return (a + b - 1) / b; }
inline int  out_dim(int H, int k, int s, int p, int d) { return (H + 2 * p - d * (k - 1) - 1) / s + 1; }

void conv_wmma(hipStream_t s, const float* x, const float* om,
               const float* w, const float* bias,
               const float* g, const float* be, const float* mn, const float* vr,
               const float* res, float* out,
               int B, int Cin, int H, int W, int Cout, int kh, int kw,
               int stride, int pad, int dil, int relu, int Ho, int Wo)
{
  int NN = B * Ho * Wo;
  dim3 grid((NN + 31) / 32, (Cout + 63) / 64);
  conv2d_wmma_kernel<<<grid, 128, 0, s>>>(x, om, w, bias, g, be, mn, vr, res, out,
                                          B, Cin, H, W, Cout, kh, kw,
                                          stride, pad, dil, Ho, Wo, relu);
}

// dcn_pack: offset conv -> fused deformable WMMA GEMM (+bn+relu)
void dcn_pack_h(hipStream_t s, Bump& ws, const float* x,
                const float* w, const float* b,
                const float* w_off, const float* b_off,
                const float* g, const float* be, const float* mn, const float* vr,
                float* out, int B, int Cin, int H, int W, int Cout,
                int k, int stride, int pad, int relu)
{
  int Ho = out_dim(H, k, stride, pad, 1);
  int Wo = out_dim(W, k, stride, pad, 1);
  size_t save = ws.off;
  float* om = ws.alloc((size_t)B * 3 * k * k * Ho * Wo);
  conv_wmma(s, x, nullptr, w_off, b_off, nullptr, nullptr, nullptr, nullptr, nullptr,
            om, B, Cin, H, W, 3 * k * k, k, k, stride, pad, 1, 0, Ho, Wo);
  conv_wmma(s, x, om, w, b, g, be, mn, vr, nullptr,
            out, B, Cin, H, W, Cout, k, k, stride, pad, 1, relu, Ho, Wo);
  ws.off = save;
}

#define PIN(i) ((const float*)d_in[(i)])

// fusion_module: returns buffer (inside caller's frame) holding fused features
float* fusion_h(hipStream_t s, Bump& ws, void* const* d_in,
                const float* fcur, const float* fprev,
                int B, int chi, int Hh, int Wh)
{
  long n0 = (long)B * chi * Hh * Wh;
  float* f = ws.alloc((size_t)n0);
  add_kernel<<<cdivl(n0, 256), 256, 0, s>>>(fcur, fprev, f, n0, 0);

  int H1 = Hh / 2, W1 = Wh / 2, H2 = H1 / 2, W2 = W1 / 2;

  // down0: chi -> 2chi, k4 s2 p1, bn+relu
  float* fd0 = ws.alloc((size_t)B * 2 * chi * H1 * W1);
  dcn_pack_h(s, ws, f, PIN(28), PIN(29), PIN(30), PIN(31),
             PIN(32), PIN(33), PIN(34), PIN(35),
             fd0, B, chi, Hh, Wh, 2 * chi, 4, 2, 1, 1);

  // down1: 2chi -> 4chi, k4 s2 p1, bn+relu
  float* fd1 = ws.alloc((size_t)B * 4 * chi * H2 * W2);
  dcn_pack_h(s, ws, fd0, PIN(40), PIN(41), PIN(42), PIN(43),
             PIN(44), PIN(45), PIN(46), PIN(47),
             fd1, B, 2 * chi, H1, W1, 4 * chi, 4, 2, 1, 1);

  // up1: upsample fd1 -> (4chi,H1,W1); dcn k3 -> 2chi, relu
  float* u1 = ws.alloc((size_t)B * 4 * chi * H1 * W1);
  { long n = (long)B * 4 * chi * H1 * W1;
    upsample2x_kernel<<<cdivl(n, 256), 256, 0, s>>>(fd1, u1, B, 4 * chi, H2, W2); }
  float* fu1 = ws.alloc((size_t)B * 2 * chi * H1 * W1);
  dcn_pack_h(s, ws, u1, PIN(48), PIN(49), PIN(50), PIN(51),
             nullptr, nullptr, nullptr, nullptr,
             fu1, B, 4 * chi, H1, W1, 2 * chi, 3, 1, 1, 1);

  // g1 = relu(fd0 + fu1), in-place fd0
  { long n = (long)B * 2 * chi * H1 * W1;
    add_kernel<<<cdivl(n, 256), 256, 0, s>>>(fd0, fu1, fd0, n, 1); }

  // up0: upsample g1 -> (2chi,Hh,Wh); dcn k3 -> chi, relu
  float* u0 = ws.alloc((size_t)B * 2 * chi * Hh * Wh);
  { long n = (long)B * 2 * chi * Hh * Wh;
    upsample2x_kernel<<<cdivl(n, 256), 256, 0, s>>>(fd0, u0, B, 2 * chi, H1, W1); }
  float* fu0 = ws.alloc((size_t)B * chi * Hh * Wh);
  dcn_pack_h(s, ws, u0, PIN(36), PIN(37), PIN(38), PIN(39),
             nullptr, nullptr, nullptr, nullptr,
             fu0, B, 2 * chi, Hh, Wh, chi, 3, 1, 1, 1);

  // result = relu(f + fu0), in-place f
  add_kernel<<<cdivl(n0, 256), 256, 0, s>>>(f, fu0, f, n0, 1);
  return f;
}

} // namespace

extern "C" void kernel_launch(void* const* d_in, const int* in_sizes, int n_in,
                              void* d_out, int out_size, void* d_ws, size_t ws_size,
                              hipStream_t stream)
{
  (void)in_sizes; (void)n_in; (void)out_size;
  const int B = 2, C = 64, H = 128, W = 128, chi = 128, Hh = 64, Wh = 64;

  const float* f_in[3] = { PIN(0), PIN(1), PIN(2) };
  Bump ws{ (char*)d_ws, 0, ws_size };

  // --- conv_in: C -> chi, k4 s2 p1, bn+relu (per frame)
  float* feats[3];
  for (int i = 0; i < 3; ++i) {
    feats[i] = ws.alloc((size_t)B * chi * Hh * Wh);
    conv_wmma(stream, f_in[i], nullptr, PIN(3), nullptr,
              PIN(4), PIN(5), PIN(6), PIN(7), nullptr,
              feats[i], B, C, H, W, chi, 4, 4, 2, 1, 1, 1, Hh, Wh);
  }

  // --- output partitions (concatenated tuple)
  float* out_main = (float*)d_out;                               // (2,64,128,128)
  float* out_off0 = out_main + (size_t)B * C * H * W;            // (2,2,64,64)
  float* out_off1 = out_off0 + (size_t)B * 2 * Hh * Wh;
  float* out_unc0 = out_off1 + (size_t)B * 2 * Hh * Wh;          // (2,1,64,64)
  float* out_unc1 = out_unc0 + (size_t)B * Hh * Wh;

  // --- main temporal loop (REPEAT=1, FRAMES=3 -> t=0,1)
  for (int t = 0; t < 2; ++t) {
    float* f_prev = feats[2 - t];
    float* f_cur  = feats[1 - t];
    size_t frame = ws.off;

    float* f_add = fusion_h(stream, ws, d_in, f_cur, f_prev, B, chi, Hh, Wh);

    // conv_offset_mask: relu(conv7x7) -> conv3x3 (27 ch)
    float* t1 = ws.alloc((size_t)B * chi * Hh * Wh);
    conv_wmma(stream, f_add, nullptr, PIN(20), PIN(21),
              nullptr, nullptr, nullptr, nullptr, nullptr,
              t1, B, chi, Hh, Wh, chi, 7, 7, 1, 3, 1, 1, Hh, Wh);
    float* om2 = ws.alloc((size_t)B * 27 * Hh * Wh);
    conv_wmma(stream, t1, nullptr, PIN(22), PIN(23),
              nullptr, nullptr, nullptr, nullptr, nullptr,
              om2, B, chi, Hh, Wh, 27, 3, 3, 1, 1, 1, 0, Hh, Wh);

    // f_align = mod_deform_conv(f_prev, om2, dcn.w, dcn.b) — fused in WMMA kernel
    float* f_align = ws.alloc((size_t)B * chi * Hh * Wh);
    conv_wmma(stream, f_prev, om2, PIN(26), PIN(27),
              nullptr, nullptr, nullptr, nullptr, nullptr,
              f_align, B, chi, Hh, Wh, chi, 3, 3, 1, 1, 1, 0, Hh, Wh);

    // alpha logits (1 channel)
    float* al = ws.alloc((size_t)B * Hh * Wh);
    conv_wmma(stream, f_add, nullptr, PIN(18), PIN(19),
              nullptr, nullptr, nullptr, nullptr, nullptr,
              al, B, chi, Hh, Wh, 1, 3, 3, 1, 1, 1, 0, Hh, Wh);

    // blend into f_cur (in place)
    { long n = (long)B * chi * Hh * Wh;
      blend_kernel<<<cdivl(n, 256), 256, 0, stream>>>(f_align, f_cur, al, B, chi, Hh, Wh); }

    // ret_off / ret_unc  (t=0 -> index 1, t=1 -> index 0)
    { long n = (long)B * 2 * Hh * Wh;
      offset_ret_kernel<<<cdivl(n, 256), 256, 0, stream>>>(
          om2, (t == 0) ? out_off1 : out_off0, B, 9, Hh, Wh); }
    conv_wmma(stream, f_add, nullptr, PIN(24), PIN(25),
              nullptr, nullptr, nullptr, nullptr, nullptr,
              (t == 0) ? out_unc1 : out_unc0,
              B, chi, Hh, Wh, 1, 3, 3, 1, 1, 1, 0, Hh, Wh);

    ws.off = frame;
  }

  // --- decoder: convT(feats[0]) -> bn1 -> conv3x3 -> bn2+relu -> +f0
  float* y1 = ws.alloc((size_t)B * C * H * W);
  { long n = (long)B * C * H * W;
    conv_transpose_kernel<<<cdivl(n, 256), 256, 0, stream>>>(
        feats[0], PIN(8), y1, B, chi, Hh, Wh, C, 4, 2, 1);
    bn_kernel<<<cdivl(n, 256), 256, 0, stream>>>(
        y1, PIN(9), PIN(10), PIN(11), PIN(12), B, C, H * W, 0); }
  conv_wmma(stream, y1, nullptr, PIN(13), nullptr,
            PIN(14), PIN(15), PIN(16), PIN(17), f_in[0],
            out_main, B, C, H, W, C, 3, 3, 1, 1, 1, 1, H, W);
}
